// PatternFFN_64656437674679
// MI455X (gfx1250) — matrix operand order repaired and checked by reading, not compiled, not run on hardware
//
#include <hip/hip_runtime.h>
#include <hip/hip_bf16.h>
#include <math.h>

// ---------------------------------------------------------------------------
// PatternFFN on MI455X (gfx1250): bf16 WMMA GEMMs, async global->LDS staging,
// fused gate/GELU epilogue.
// ---------------------------------------------------------------------------

typedef __bf16 v16bf __attribute__((ext_vector_type(16)));
typedef __bf16 v8bf  __attribute__((ext_vector_type(8)));
typedef float  v8f   __attribute__((ext_vector_type(8)));

#define BS_TOK   8192      // B*S
#define D_MODEL_ 1024
#define D_FF_    4096
#define N_PAT    64
#define K_PAT    4
#define K_NEUR   8

__device__ __forceinline__ unsigned short f32_to_bf16_rne(float f) {
    unsigned u = __float_as_uint(f);
    u += 0x7FFFu + ((u >> 16) & 1u);       // round-to-nearest-even
    return (unsigned short)(u >> 16);
}

// Async 16B global -> LDS copy (gfx1250 GLOBAL_LOAD_ASYNC_TO_LDS_B128).
// ldsOfs: LDS byte offset (low 32 bits of a generic shared pointer, per the
// ISA aperture rule); gaddr: 64-bit global address.
__device__ __forceinline__ void async_copy_b128(unsigned ldsOfs, unsigned long long gaddr) {
    asm volatile("global_load_async_to_lds_b128 %0, %1, off"
                 :: "v"(ldsOfs), "v"(gaddr) : "memory");
}
__device__ __forceinline__ void wait_async0() {
    asm volatile("s_wait_asynccnt 0x0" ::: "memory");
}

// --------------------------- f32 -> bf16 cast ------------------------------
__global__ __launch_bounds__(256)
void cvt_bf16_kernel(const float* __restrict__ in, unsigned short* __restrict__ out, int n) {
    int i = blockIdx.x * blockDim.x + threadIdx.x;
    int stride = gridDim.x * blockDim.x;
    for (; i < n; i += stride) out[i] = f32_to_bf16_rne(in[i]);
}

// ------------------- pattern scoring / top-4 / softmax ----------------------
__global__ __launch_bounds__(256)
void score_kernel(const float* __restrict__ sel, const float* __restrict__ tw,
                  const float* __restrict__ pq, int* __restrict__ pidx,
                  float* __restrict__ pw) {
    __shared__ float vsh[D_MODEL_];
    __shared__ float ssh[N_PAT];
    const int token = blockIdx.x;
    const int tid = threadIdx.x;
    const float* selt = sel + (size_t)token * K_NEUR * D_MODEL_;
    float w[K_NEUR];
#pragma unroll
    for (int k = 0; k < K_NEUR; ++k) w[k] = tw[token * K_NEUR + k];
    float4 v = make_float4(0.f, 0.f, 0.f, 0.f);
#pragma unroll
    for (int k = 0; k < K_NEUR; ++k) {
        float4 s = reinterpret_cast<const float4*>(selt + k * D_MODEL_)[tid];
        v.x += w[k] * s.x; v.y += w[k] * s.y; v.z += w[k] * s.z; v.w += w[k] * s.w;
    }
    reinterpret_cast<float4*>(vsh)[tid] = v;
    __syncthreads();

    const int g = tid >> 2, sub = tid & 3;           // 64 groups of 4 lanes
    const float* pqg = pq + g * D_MODEL_;
    float s = 0.f;
    for (int d = sub; d < D_MODEL_; d += 4) s += vsh[d] * pqg[d];
    s += __shfl_xor(s, 1, 32);
    s += __shfl_xor(s, 2, 32);
    if (sub == 0) ssh[g] = s * 0.03125f;             // 1/sqrt(1024)
    __syncthreads();

    if (tid == 0) {
        float bs[K_PAT]; int bi[K_PAT];
        for (int j = 0; j < K_PAT; ++j) {
            float best = -3.4e38f; int bidx = 0;
            for (int p = 0; p < N_PAT; ++p) {
                float sv = ssh[p];
                if (sv > best) { best = sv; bidx = p; }
            }
            ssh[bidx] = -3.4e38f;
            bs[j] = best; bi[j] = bidx;
        }
        float e[K_PAT], sum = 0.f;
        for (int j = 0; j < K_PAT; ++j) { e[j] = __expf(bs[j] - bs[0]); sum += e[j]; }
        float inv = 1.f / sum;
        for (int j = 0; j < K_PAT; ++j) {
            pidx[token * K_PAT + j] = bi[j];
            pw[token * K_PAT + j] = e[j] * inv;
        }
    }
}

// ------------------------------- WMMA GEMM ---------------------------------
// C[M,N] = A[M,K] * Bw[N,K]^T (both bf16 row-major), f32 accumulate.
// Block: 256 threads (8 waves), 128x128 tile, K-step 32, double-buffered LDS
// filled by async global->LDS B128 copies (ASYNCcnt).
// Wave grid 2x4, each wave -> 64x32 output = 4x2 v_wmma_f32_16x16x32_bf16.
// EPI==1: fused +up_b, gate gather (4 patterns), sigmoid, exact GELU -> bf16.
// EPI==2: +down_b -> f32.
template <int EPI>
__global__ __launch_bounds__(256)
void gemm_kernel(const unsigned short* __restrict__ A,
                 const unsigned short* __restrict__ Bw,
                 float* __restrict__ outF32, unsigned short* __restrict__ outBf16,
                 const float* __restrict__ bias,
                 const float* __restrict__ gates, const int* __restrict__ pidx,
                 const float* __restrict__ pw,
                 int M, int N, int Kdim) {
    constexpr int TILE = 128, KS = 32, LDSS = 40;  // padded stride: bank-conflict free
    __shared__ __bf16 As[2][TILE * LDSS];
    __shared__ __bf16 Bs[2][TILE * LDSS];

    const int tid = threadIdx.x;
    const int lane = tid & 31;
    const int wv = tid >> 5;
    const int waveM = wv >> 2;                     // 0..1
    const int waveN = wv & 3;                      // 0..3
    const int rowBase = blockIdx.x * TILE;
    const int colBase = blockIdx.y * TILE;

    // Issue async copies for one 128x32 A tile + 128x32 B tile into buffer buf.
    auto loadTileAsync = [&](int buf, int k0) {
#pragma unroll
        for (int it = 0; it < 2; ++it) {
            int c = tid + it * 256;                // 512 chunks of 8 bf16
            int r = c >> 2, cc = (c & 3) * 8;
            unsigned lofs = (unsigned)(unsigned long long)(const void*)&As[buf][r * LDSS + cc];
            unsigned long long g =
                (unsigned long long)(A + (size_t)(rowBase + r) * Kdim + k0 + cc);
            async_copy_b128(lofs, g);
        }
#pragma unroll
        for (int it = 0; it < 2; ++it) {
            int c = tid + it * 256;
            int r = c >> 2, cc = (c & 3) * 8;
            unsigned lofs = (unsigned)(unsigned long long)(const void*)&Bs[buf][r * LDSS + cc];
            unsigned long long g =
                (unsigned long long)(Bw + (size_t)(colBase + r) * Kdim + k0 + cc);
            async_copy_b128(lofs, g);
        }
    };

    v8f acc[4][2] = {};
    const int nk = Kdim / KS;
    loadTileAsync(0, 0);
    const int l15 = lane & 15;
    const int acb = (lane < 16) ? 0 : 8;           // A: K chunk base (ISA 16-bit A layout)
    const int bkb = (lane < 16) ? 0 : 8;           // B: K base (ISA 16-bit B layout)

    for (int ks = 0; ks < nk; ++ks) {
        wait_async0();                             // my async copies landed in LDS
        __syncthreads();                           // everyone's copies visible
        if (ks + 1 < nk) loadTileAsync((ks + 1) & 1, (ks + 1) * KS);
        const int buf = ks & 1;
        v16bf afr[4], bfr[2];
#pragma unroll
        for (int i = 0; i < 4; ++i) {
            int r = waveM * 64 + i * 16 + l15;
            v8bf lo = *reinterpret_cast<const v8bf*>(&As[buf][r * LDSS + acb]);
            v8bf hi = *reinterpret_cast<const v8bf*>(&As[buf][r * LDSS + acb + 16]);
            afr[i] = __builtin_shufflevector(lo, hi, 0, 1, 2, 3, 4, 5, 6, 7,
                                             8, 9, 10, 11, 12, 13, 14, 15);
        }
#pragma unroll
        for (int j = 0; j < 2; ++j) {
            int ccol = waveN * 32 + j * 16 + l15;
            v8bf lo = *reinterpret_cast<const v8bf*>(&Bs[buf][ccol * LDSS + bkb]);
            v8bf hi = *reinterpret_cast<const v8bf*>(&Bs[buf][ccol * LDSS + bkb + 16]);
            bfr[j] = __builtin_shufflevector(lo, hi, 0, 1, 2, 3, 4, 5, 6, 7,
                                             8, 9, 10, 11, 12, 13, 14, 15);
        }
#pragma unroll
        for (int i = 0; i < 4; ++i)
#pragma unroll
            for (int j = 0; j < 2; ++j)
                acc[i][j] = __builtin_amdgcn_wmma_f32_16x16x32_bf16(
                    false, afr[i], false, bfr[j], (short)0, acc[i][j], false, false);
    }

    // Epilogue. C layout: lane<16 -> N=lane, M=e; lane>=16 -> N=lane-16, M=8+e.
    const int mofs = (lane < 16) ? 0 : 8;
#pragma unroll
    for (int i = 0; i < 4; ++i)
#pragma unroll
        for (int j = 0; j < 2; ++j) {
            const int n = colBase + waveN * 32 + j * 16 + l15;
            const float bn = bias[n];
#pragma unroll
            for (int e = 0; e < 8; ++e) {
                const int m = rowBase + waveM * 64 + i * 16 + e + mofs;
                float v = acc[i][j][e] + bn;
                if (EPI == 1) {
                    float gate = 0.f;
#pragma unroll
                    for (int q = 0; q < K_PAT; ++q)
                        gate += pw[m * K_PAT + q] *
                                gates[(size_t)pidx[m * K_PAT + q] * D_FF_ + n];
                    float gx = v / (1.f + __expf(-gate));            // h * sigmoid(gate)
                    float r = 0.5f * gx * (1.f + erff(gx * 0.70710678118654752f));
                    outBf16[(size_t)m * N + n] = f32_to_bf16_rne(r);
                } else {
                    outF32[(size_t)m * N + n] = v;
                }
            }
        }
}

// ------------------------------- launcher ----------------------------------
extern "C" void kernel_launch(void* const* d_in, const int* in_sizes, int n_in,
                              void* d_out, int out_size, void* d_ws, size_t ws_size,
                              hipStream_t stream) {
    const float* x      = (const float*)d_in[0];
    // d_in[1] router_out, d_in[2] topk_neuron_idx: unused by reference
    const float* tw     = (const float*)d_in[3];
    const float* sel    = (const float*)d_in[4];
    const float* pq     = (const float*)d_in[5];
    const float* gates  = (const float*)d_in[6];
    const float* up_w   = (const float*)d_in[7];
    const float* up_b   = (const float*)d_in[8];
    const float* down_w = (const float*)d_in[9];
    const float* down_b = (const float*)d_in[10];
    float* out = (float*)d_out;

    char* ws = (char*)d_ws;
    unsigned short* xb  = (unsigned short*)ws; ws += (size_t)BS_TOK * D_MODEL_ * 2;
    unsigned short* uwb = (unsigned short*)ws; ws += (size_t)D_FF_ * D_MODEL_ * 2;
    unsigned short* dwb = (unsigned short*)ws; ws += (size_t)D_MODEL_ * D_FF_ * 2;
    unsigned short* hb  = (unsigned short*)ws; ws += (size_t)BS_TOK * D_FF_ * 2;
    int*   pidx = (int*)ws;                    ws += (size_t)BS_TOK * K_PAT * 4;
    float* pwts = (float*)ws;

    cvt_bf16_kernel<<<2048, 256, 0, stream>>>(x, xb, BS_TOK * D_MODEL_);
    cvt_bf16_kernel<<<2048, 256, 0, stream>>>(up_w, uwb, D_FF_ * D_MODEL_);
    cvt_bf16_kernel<<<2048, 256, 0, stream>>>(down_w, dwb, D_MODEL_ * D_FF_);

    score_kernel<<<BS_TOK, 256, 0, stream>>>(sel, tw, pq, pidx, pwts);

    gemm_kernel<1><<<dim3(BS_TOK / 128, D_FF_ / 128), 256, 0, stream>>>(
        xb, uwb, nullptr, hb, up_b, gates, pidx, pwts, BS_TOK, D_FF_, D_MODEL_);

    gemm_kernel<2><<<dim3(BS_TOK / 128, D_MODEL_ / 128), 256, 0, stream>>>(
        hb, dwb, out, nullptr, down_b, nullptr, nullptr, nullptr, BS_TOK, D_MODEL_, D_FF_);
}
